// SoftDTW_31353261261227
// MI455X (gfx1250) — matrix operand order repaired and checked by reading, not compiled, not run on hardware
//
#include <hip/hip_runtime.h>
#include <stdint.h>
#include <math.h>

typedef __attribute__((ext_vector_type(16))) _Float16 v16h;
typedef __attribute__((ext_vector_type(8)))  _Float16 v8h;
typedef __attribute__((ext_vector_type(8)))  float    v8f;

#define MDIM 2048
#define NDIM 2048
#define DDIM 1024
#define BIGF 1e9f

// GEMM macro-tiling
#define BM 128
#define BN 128
#define KC 32
#define NCHUNK (DDIM / KC)   // 32
#define APAD 8
#define ASTR (KC + APAD)     // 40 halves = 80 B row stride (conflict-free b128 reads)

// ---------------- 1) Row L2-normalize f32 -> f16 ----------------
__global__ void norm_rows_kernel(const float* __restrict__ in,
                                 _Float16* __restrict__ out) {
  __shared__ float red[256];
  const int row = blockIdx.x;
  const int tid = threadIdx.x;
  const float* r = in + (size_t)row * DDIM;
  float ss = 0.f;
  for (int i = tid; i < DDIM; i += 256) { float v = r[i]; ss += v * v; }
  red[tid] = ss;
  __syncthreads();
  for (int s = 128; s > 0; s >>= 1) {
    if (tid < s) red[tid] += red[tid + s];
    __syncthreads();
  }
  const float scale = 1.0f / fmaxf(sqrtf(red[0]), 1e-12f);
  _Float16* o = out + (size_t)row * DDIM;
  for (int i = tid; i < DDIM; i += 256) o[i] = (_Float16)(r[i] * scale);
}

// ---------------- 2) cost = 1 - An * Bn^T (LDS-tiled WMMA) ----------------
// 256 threads = 8 waves compute a 128x128 tile. Waves in a 4(M) x 2(N) grid;
// each wave owns 2x4 16x16 WMMA tiles. A/B 128x32 panels are staged to LDS
// with gfx1250 async LDS loads (ASYNCcnt), double-buffered.
__global__ void __launch_bounds__(256)
cost_wmma_kernel(const _Float16* __restrict__ A,
                 const _Float16* __restrict__ B,
                 float* __restrict__ C) {
  __shared__ _Float16 As[2][BM * ASTR];
  __shared__ _Float16 Bs[2][BN * ASTR];

  const int tid  = threadIdx.x;
  const int lane = tid & 31;
  const int wave = tid >> 5;
  const int wm   = wave & 3;        // wave row   (0..3) -> 32 rows each
  const int wn   = wave >> 2;       // wave col   (0..1) -> 64 cols each
  const int bm0  = blockIdx.x * BM;
  const int bn0  = blockIdx.y * BN;

  // staging map: thread t moves 16 halves (32B) of A and of B per chunk
  const int srow = tid >> 1;            // 0..127
  const int skp  = (tid & 1) * 16;      // 0 or 16
  const _Float16* ag = A + (size_t)(bm0 + srow) * DDIM + skp;
  const _Float16* bg = B + (size_t)(bn0 + srow) * DDIM + skp;
  const unsigned sdst = (unsigned)(srow * ASTR + skp) * 2u;  // byte offset in panel

  // WMMA fragment lane mapping (ISA 7.12.2, 16-bit operands)
  const int lrow  = lane & 15;
  const int ahalf = (lane >> 4) * 8;
  const int bhalf = (lane >> 4) * 8;

  v8f acc[2][4];
#pragma unroll
  for (int tm = 0; tm < 2; ++tm)
#pragma unroll
    for (int tn = 0; tn < 4; ++tn) acc[tm][tn] = (v8f){};

  // issue 4 async global->LDS b128 copies (2 for A panel, 2 for B panel)
  auto stage = [&](int buf, int k0) {
    unsigned la = (unsigned)(uintptr_t)(&As[buf][0]) + sdst;
    unsigned lb = (unsigned)(uintptr_t)(&Bs[buf][0]) + sdst;
    unsigned long long ga = (unsigned long long)(uintptr_t)(ag + k0);
    unsigned long long gb = (unsigned long long)(uintptr_t)(bg + k0);
    asm volatile("global_load_async_to_lds_b128 %0, %1, off"
                 :: "v"(la),        "v"(ga)         : "memory");
    asm volatile("global_load_async_to_lds_b128 %0, %1, off"
                 :: "v"(la + 16u),  "v"(ga + 16ull) : "memory");
    asm volatile("global_load_async_to_lds_b128 %0, %1, off"
                 :: "v"(lb),        "v"(gb)         : "memory");
    asm volatile("global_load_async_to_lds_b128 %0, %1, off"
                 :: "v"(lb + 16u),  "v"(gb + 16ull) : "memory");
  };

  stage(0, 0);
  int buf = 0;
  for (int c = 0; c < NCHUNK; ++c) {
    if (c + 1 < NCHUNK) {
      stage(buf ^ 1, (c + 1) * KC);               // prefetch next chunk
      asm volatile("s_wait_asynccnt 0x4" ::: "memory");  // current chunk done
    } else {
      asm volatile("s_wait_asynccnt 0x0" ::: "memory");
    }
    __syncthreads();

    v16h af[2], bf[4];
#pragma unroll
    for (int tm = 0; tm < 2; ++tm) {
      const int r = wm * 32 + tm * 16 + lrow;
      v8h lo = *(const v8h*)&As[buf][r * ASTR + ahalf];
      v8h hi = *(const v8h*)&As[buf][r * ASTR + 16 + ahalf];
      af[tm] = __builtin_shufflevector(lo, hi,
          0, 1, 2, 3, 4, 5, 6, 7, 8, 9, 10, 11, 12, 13, 14, 15);
    }
#pragma unroll
    for (int tn = 0; tn < 4; ++tn) {
      const int r = wn * 64 + tn * 16 + lrow;
      v8h lo = *(const v8h*)&Bs[buf][r * ASTR + bhalf];
      v8h hi = *(const v8h*)&Bs[buf][r * ASTR + 16 + bhalf];
      bf[tn] = __builtin_shufflevector(lo, hi,
          0, 1, 2, 3, 4, 5, 6, 7, 8, 9, 10, 11, 12, 13, 14, 15);
    }
#pragma unroll
    for (int tm = 0; tm < 2; ++tm)
#pragma unroll
      for (int tn = 0; tn < 4; ++tn)
        acc[tm][tn] = __builtin_amdgcn_wmma_f32_16x16x32_f16(
            false, af[tm], false, bf[tn], (short)0, acc[tm][tn], false, false);

    __syncthreads();   // all waves done reading `buf` before it is re-staged
    buf ^= 1;
  }

  // store: C/D layout -> VGPR rr = row 8*(lane>=16)+rr, col lane&15
  const int mb = (lane >> 4) * 8;
  const int nn = lane & 15;
#pragma unroll
  for (int tm = 0; tm < 2; ++tm) {
#pragma unroll
    for (int tn = 0; tn < 4; ++tn) {
      float* cbase = C + (size_t)(bm0 + wm * 32 + tm * 16 + mb) * NDIM
                       + (size_t)(bn0 + wn * 64 + tn * 16 + nn);
#pragma unroll
      for (int rr = 0; rr < 8; ++rr)
        cbase[(size_t)rr * NDIM] = 1.0f - acc[tm][tn][rr];
    }
  }
}

// ---------------- 3) Soft-DTW anti-diagonal wavefront DP ----------------
__global__ void __launch_bounds__(1024)
softdtw_kernel(const float* __restrict__ costs,
               const float* __restrict__ gamma,
               float* __restrict__ results) {
  __shared__ float R[3][MDIM + 16];
  const float* cost = costs + (size_t)blockIdx.x * MDIM * NDIM;
  const int tid = threadIdx.x;
  const float g = fmaxf(fabsf(gamma[0]), 1e-4f);
  const float invg = 1.0f / g;

  for (int i = tid; i <= MDIM; i += 1024) {
    R[0][i] = BIGF; R[1][i] = BIGF; R[2][i] = BIGF;
  }
  __syncthreads();
  if (tid == 0) R[0][0] = 0.0f;
  __syncthreads();

  int cur = 2, p1 = 1, p2 = 0;
  for (int d = 2; d <= MDIM + NDIM; ++d) {
#pragma unroll
    for (int rep = 0; rep < 3; ++rep) {
      const int i = tid + rep * 1024;
      if (i <= MDIM) {
        const int j = d - i;
        float val = BIGF;
        if (i >= 1 && j >= 1 && j <= NDIM) {
          const float c  = cost[(size_t)(i - 1) * NDIM + (j - 1)];
          const float dv = R[p2][i - 1];
          const float av = R[p1][i - 1];
          const float lv = R[p1][i];
          const float mn = fminf(dv, fminf(av, lv));
          const float s  = __expf((mn - dv) * invg)
                         + __expf((mn - av) * invg)
                         + __expf((mn - lv) * invg);
          val = c + mn - g * __logf(s);
        }
        R[cur][i] = val;
      }
    }
    __syncthreads();
    const int t = p2; p2 = p1; p1 = cur; cur = t;
  }
  if (tid == 0) results[blockIdx.x] = R[p1][MDIM];
}

// ---------------- 4) out = sdtw_xy - 0.5*(sdtw_xx + sdtw_yy) ----------------
__global__ void combine_kernel(const float* __restrict__ results,
                               float* __restrict__ out) {
  out[0] = results[0] - 0.5f * (results[1] + results[2]);
}

extern "C" void kernel_launch(void* const* d_in, const int* in_sizes, int n_in,
                              void* d_out, int out_size, void* d_ws, size_t ws_size,
                              hipStream_t stream) {
  (void)in_sizes; (void)n_in; (void)out_size; (void)ws_size;
  const float* x = (const float*)d_in[0];
  const float* y = (const float*)d_in[1];
  const float* gammap = (const float*)d_in[2];

  char* ws = (char*)d_ws;
  const size_t f16_mat  = (size_t)MDIM * DDIM * sizeof(_Float16);  // 4 MiB
  const size_t cost_mat = (size_t)MDIM * NDIM * sizeof(float);     // 16 MiB
  _Float16* xb = (_Float16*)ws;
  _Float16* yb = (_Float16*)(ws + f16_mat);
  float* costs   = (float*)(ws + 2 * f16_mat);                     // 3 matrices
  float* results = (float*)(ws + 2 * f16_mat + 3 * cost_mat);

  norm_rows_kernel<<<MDIM, 256, 0, stream>>>(x, xb);
  norm_rows_kernel<<<NDIM, 256, 0, stream>>>(y, yb);

  dim3 gg(MDIM / BM, NDIM / BN);   // 16 x 16 blocks per matrix
  cost_wmma_kernel<<<gg, 256, 0, stream>>>(xb, yb, costs);
  cost_wmma_kernel<<<gg, 256, 0, stream>>>(xb, xb, costs + (size_t)MDIM * NDIM);
  cost_wmma_kernel<<<gg, 256, 0, stream>>>(yb, yb, costs + (size_t)2 * MDIM * NDIM);

  softdtw_kernel<<<3, 1024, 0, stream>>>(costs, gammap, results);

  combine_kernel<<<1, 1, 0, stream>>>(results, (float*)d_out);
}
